// BaseBox2dHead_52699248722265
// MI455X (gfx1250) — compile-verified, matrix operand order checked by reading, not run on hardware
//
#include <hip/hip_runtime.h>
#include <hip/hip_fp16.h>

// ---------------- problem constants (from reference) ----------------
#define B_IMG       16
#define N_FEAT      16384
#define CIN         256
#define NUM_LABELS  81
#define NUM_CLASSES 80
#define KNMS        1000      // NMS_CAND
#define MAX_DETS    100
#define NMS_THR     0.5f
#define BBOX_CLIP   4.135166556742356f   // log(1000/16)

// ---------------- pipeline tuning ----------------
#define NB   4096             // histogram buckets over logit range [-16, 16]
#define CAP  2048             // per-image candidate capacity (expect ~1030)
#define NEG_INF (-__builtin_inff())

typedef __attribute__((ext_vector_type(16))) _Float16 v16h;
typedef __attribute__((ext_vector_type(8)))  float    v8f;

// =====================================================================
// Kernel 1: box-logit GEMM (WMMA f16->f32) + box delta decode.
// One wave32 per 16-row tile; block = 256 threads = 8 waves = 128 rows.
// Memory bound: streams in_feats (268 MB) once. B fragments (from the
// loop-invariant 256x4 W) are hoisted to registers; the K loop is fully
// unrolled into 8 back-to-back v_wmma so A loads pipeline across them.
// =====================================================================
__global__ __launch_bounds__(256)
void gemm_decode_kernel(const float* __restrict__ feats,
                        const float* __restrict__ Wm,     // [256,4] row-major
                        const float* __restrict__ bias,   // [4]
                        const float* __restrict__ priors, // [B*N,4] cxcywh
                        float* __restrict__ pred)         // [B*N,4] cxcywh out
{
  __shared__ float s_logit[8][16][4];
  const int tid  = threadIdx.x;
  const int wave = tid >> 5;          // wave32
  const int lane = tid & 31;
  const int m    = lane & 15;         // row within tile (A) / column n (B,D)
  const int h    = lane >> 4;         // lane half selects K sub-range
  const int row0 = blockIdx.x * 128 + wave * 16;
  const float* arow = feats + (size_t)(row0 + m) * CIN;
  const int   ncol  = m & 3;
  const float nmask = (m < 4) ? 1.0f : 0.0f;  // only N=0..3 columns are real

  // -------- hoist the 8 loop-invariant B fragments into registers --------
  // 16-bit B 32x16 layout (symmetric to A): lane holds column n = lane%16;
  // elements 0..7 -> K = k0+h*8..+7, elements 8..15 -> K = k0+16+h*8..+7.
  v16h bf[8];
#pragma unroll
  for (int kc = 0; kc < 8; ++kc) {
    const int k0 = kc * 32;
#pragma unroll
    for (int e = 0; e < 8; ++e) {
      const int kA = k0 + h * 8 + e;
      const int kB = kA + 16;
      bf[kc][e]     = (_Float16)(Wm[kA * 4 + ncol] * nmask);
      bf[kc][e + 8] = (_Float16)(Wm[kB * 4 + ncol] * nmask);
    }
  }

  v8f acc = {};
#pragma unroll
  for (int kc = 0; kc < 8; ++kc) {
    const int k0 = kc * 32;
    // A fragment: 16-bit A 16x32 layout; two contiguous 8-float runs per lane.
    v16h a;
    const float* p0 = arow + k0 + h * 8;
    const float* p1 = p0 + 16;
#pragma unroll
    for (int e = 0; e < 8; ++e) {
      a[e]     = (_Float16)p0[e];
      a[e + 8] = (_Float16)p1[e];
    }
    acc = __builtin_amdgcn_wmma_f32_16x16x32_f16(false, a, false, bf[kc],
                                                 (short)0, acc, false, false);
  }

  // D layout: lane holds column n=lane%16, rows m = v + 8*(lane/16), v=0..7.
  if (m < 4) {
#pragma unroll
    for (int v = 0; v < 8; ++v)
      s_logit[wave][v + 8 * h][m] = acc[v];
  }
  __syncthreads();

  // decode one row per thread (threads 0..127). tid == wavetile*16 + r.
  if (tid < 128) {
    const int w2 = tid >> 4, r = tid & 15;
    const size_t row = (size_t)blockIdx.x * 128 + tid;
    const float dx = s_logit[w2][r][0] + bias[0];
    const float dy = s_logit[w2][r][1] + bias[1];
    const float dw = s_logit[w2][r][2] + bias[2];
    const float dh = s_logit[w2][r][3] + bias[3];
    const float4 pr = ((const float4*)priors)[row];
    float4 o;
    o.x = pr.x + dx * pr.z;
    o.y = pr.y + dy * pr.w;
    o.z = pr.z * __expf(fminf(dw, BBOX_CLIP));
    o.w = pr.w * __expf(fminf(dh, BBOX_CLIP));
    ((float4*)pred)[row] = o;
  }
}

// =====================================================================
__device__ __forceinline__ int bucket_of(float x) {
  int b = (int)((x + 16.0f) * (NB / 32.0f));
  return b < 0 ? 0 : (b >= NB ? NB - 1 : b);
}

__global__ __launch_bounds__(256)
void init_ws_kernel(int* __restrict__ hist, int* __restrict__ thrb,
                    int* __restrict__ cnt, float* __restrict__ csc,
                    int* __restrict__ cix) {
  const int i = blockIdx.x * blockDim.x + threadIdx.x;
  if (i < B_IMG * NB)  hist[i] = 0;
  if (i < B_IMG)       { thrb[i] = 0; cnt[i] = 0; }
  if (i < B_IMG * CAP) { csc[i] = NEG_INF; cix[i] = 0x7fffffff; }
}

// Kernel 2: per-image logit histograms. 64 blocks per image, 256 feats each.
__global__ __launch_bounds__(256)
void hist_kernel(const float* __restrict__ clog, const float* __restrict__ pred,
                 int* __restrict__ hist) {
  __shared__ int sh[NB];
  const int tid = threadIdx.x;
  const int img = blockIdx.x >> 6;
  const int chunk = blockIdx.x & 63;
  for (int i = tid; i < NB; i += 256) sh[i] = 0;
  __syncthreads();
  const int feat = chunk * 256 + tid;
  const size_t row = (size_t)img * N_FEAT + feat;
  const float2 wh = ((const float2*)pred)[row * 2 + 1];   // (w, h)
  if (wh.x > 0.0f && wh.y > 0.0f) {                       // well_defined
    const float* lrow = clog + row * NUM_LABELS;
    for (int c = 0; c < NUM_CLASSES; ++c)
      atomicAdd(&sh[bucket_of(lrow[c])], 1);
  }
  __syncthreads();
  for (int i = tid; i < NB; i += 256)
    if (sh[i]) atomicAdd(&hist[img * NB + i], sh[i]);
}

// Kernel 3: threshold bucket so that cum-count-from-top >= KNMS.
__global__ void thresh_kernel(const int* __restrict__ hist, int* __restrict__ thrb) {
  const int b = threadIdx.x;
  if (b >= B_IMG) return;
  int cum = 0, t = 0;
  for (int i = NB - 1; i >= 0; --i) {
    cum += hist[b * NB + i];
    if (cum >= KNMS) { t = i; break; }
  }
  thrb[b] = t;
}

// Kernel 4: compact candidates above threshold (deterministic set; order fixed
// later by a total-order sort).
__global__ __launch_bounds__(256)
void compact_kernel(const float* __restrict__ clog, const float* __restrict__ pred,
                    const int* __restrict__ thrb, int* __restrict__ cnt,
                    float* __restrict__ csc, int* __restrict__ cix) {
  const int tid = threadIdx.x;
  const int img = blockIdx.x >> 6;
  const int chunk = blockIdx.x & 63;
  const int feat = chunk * 256 + tid;
  const size_t row = (size_t)img * N_FEAT + feat;
  const float2 wh = ((const float2*)pred)[row * 2 + 1];   // (w, h)
  const int t = thrb[img];
  if (wh.x > 0.0f && wh.y > 0.0f) {
    const float* lrow = clog + row * NUM_LABELS;
    for (int c = 0; c < NUM_CLASSES; ++c) {
      const float s = lrow[c];
      if (bucket_of(s) >= t) {
        const int pos = atomicAdd(&cnt[img], 1);
        if (pos < CAP) {
          csc[img * CAP + pos] = s;
          cix[img * CAP + pos] = feat * NUM_CLASSES + c;
        }
      }
    }
  }
}

// =====================================================================
// Kernel 5: per-image bitonic sort (desc score, asc idx), class-aware NMS,
// final top-100. One 1024-thread block per image, everything in LDS (~48KB).
// =====================================================================
__global__ __launch_bounds__(1024)
void detect_kernel(const float* __restrict__ pred,
                   const float* __restrict__ csc, const int* __restrict__ cix,
                   float* __restrict__ o_lab, float* __restrict__ o_box,
                   float* __restrict__ o_sc,  float* __restrict__ o_bi,
                   float* __restrict__ o_va) {
  __shared__ float s_sc[CAP];
  __shared__ int   s_ix[CAP];
  __shared__ float s_box[KNMS * 4];
  __shared__ float s_off[KNMS];
  __shared__ float s_area[KNMS];
  __shared__ int   s_keep[KNMS];
  __shared__ float s_red[1024];
  __shared__ int   s_map[MAX_DETS];
  const int tid = threadIdx.x;
  const int img = blockIdx.x;

  for (int i = tid; i < CAP; i += blockDim.x) {
    s_sc[i] = csc[img * CAP + i];
    s_ix[i] = cix[img * CAP + i];
  }
  // bitonic sort: descending by score, ties broken by ascending flat index
  for (int k = 2; k <= CAP; k <<= 1) {
    for (int j = k >> 1; j > 0; j >>= 1) {
      __syncthreads();
      for (int i = tid; i < CAP; i += blockDim.x) {
        const int l = i ^ j;
        if (l > i) {
          const float si = s_sc[i], sl = s_sc[l];
          const int   ii = s_ix[i], il = s_ix[l];
          const bool before = (si > sl) || (si == sl && ii < il);
          const bool up = ((i & k) == 0);
          if (up ? !before : before) {
            s_sc[i] = sl; s_sc[l] = si;
            s_ix[i] = il; s_ix[l] = ii;
          }
        }
      }
    }
  }
  __syncthreads();

  // build top-KNMS candidate boxes (xyxy) from decoded cxcywh
  for (int t = tid; t < KNMS; t += blockDim.x) {
    const float sc = s_sc[t];
    const bool  val = sc > -1e30f;
    const int   ix = s_ix[t];
    const int   feat = val ? (ix / NUM_CLASSES) : 0;
    const float4 pb = ((const float4*)pred)[(size_t)img * N_FEAT + feat];
    const float x1 = pb.x - 0.5f * pb.z, y1 = pb.y - 0.5f * pb.w;
    const float x2 = pb.x + 0.5f * pb.z, y2 = pb.y + 0.5f * pb.w;
    s_box[t * 4 + 0] = x1; s_box[t * 4 + 1] = y1;
    s_box[t * 4 + 2] = x2; s_box[t * 4 + 3] = y2;
    s_area[t] = fmaxf(x2 - x1, 0.f) * fmaxf(y2 - y1, 0.f);
    s_keep[t] = val ? 1 : 0;
  }
  __syncthreads();

  // max_c = max over where(valid, coords, 0)  (reference semantics)
  float lmax = NEG_INF;
  for (int t = tid; t < KNMS; t += blockDim.x) {
    const bool val = s_keep[t] != 0;
#pragma unroll
    for (int c2 = 0; c2 < 4; ++c2)
      lmax = fmaxf(lmax, val ? s_box[t * 4 + c2] : 0.0f);
  }
  s_red[tid] = lmax;
  __syncthreads();
  for (int s = 512; s > 0; s >>= 1) {
    if (tid < s) s_red[tid] = fmaxf(s_red[tid], s_red[tid + s]);
    __syncthreads();
  }
  const float maxc = s_red[0];
  __syncthreads();

  for (int t = tid; t < KNMS; t += blockDim.x)
    s_off[t] = (float)(s_ix[t] % NUM_CLASSES) * (maxc + 1.0f);
  __syncthreads();

  // sequential class-aware NMS (suppress j>i if IoU(offset boxes) > thr)
  for (int i = 0; i < KNMS - 1; ++i) {
    if (s_keep[i]) {
      const float oi = s_off[i];
      const float ax1 = s_box[i * 4 + 0] + oi, ay1 = s_box[i * 4 + 1] + oi;
      const float ax2 = s_box[i * 4 + 2] + oi, ay2 = s_box[i * 4 + 3] + oi;
      const float aar = s_area[i];
      for (int jj = i + 1 + tid; jj < KNMS; jj += blockDim.x) {
        if (s_keep[jj]) {
          const float oj = s_off[jj];
          const float bx1 = s_box[jj * 4 + 0] + oj, by1 = s_box[jj * 4 + 1] + oj;
          const float bx2 = s_box[jj * 4 + 2] + oj, by2 = s_box[jj * 4 + 3] + oj;
          const float iw = fmaxf(fminf(ax2, bx2) - fmaxf(ax1, bx1), 0.f);
          const float ih = fmaxf(fminf(ay2, by2) - fmaxf(ay1, by1), 0.f);
          const float inter = iw * ih;
          const float uni = aar + s_area[jj] - inter;
          const float iou = uni > 0.f ? inter / uni : 0.f;
          if (iou > NMS_THR) s_keep[jj] = 0;
        }
      }
    }
    __syncthreads();
  }

  // final top-100: kept candidates are already score-sorted
  if (tid == 0) {
    int slot = 0;
    for (int t2 = 0; t2 < KNMS && slot < MAX_DETS; ++t2)
      if (s_keep[t2]) s_map[slot++] = t2;
    for (; slot < MAX_DETS; ++slot) s_map[slot] = -1;
  }
  __syncthreads();
  if (tid < MAX_DETS) {
    const size_t o = (size_t)img * MAX_DETS + tid;
    const int t2 = s_map[tid];
    o_bi[o] = (float)img;
    if (t2 >= 0) {
      const int ix = s_ix[t2];
      o_lab[o] = (float)(ix % NUM_CLASSES);
      o_box[o * 4 + 0] = s_box[t2 * 4 + 0];
      o_box[o * 4 + 1] = s_box[t2 * 4 + 1];
      o_box[o * 4 + 2] = s_box[t2 * 4 + 2];
      o_box[o * 4 + 3] = s_box[t2 * 4 + 3];
      o_sc[o] = 1.0f / (1.0f + __expf(-s_sc[t2]));  // sigmoid at the end
      o_va[o] = 1.0f;
    } else {
      o_lab[o] = -1.0f;
      o_box[o * 4 + 0] = 0.f; o_box[o * 4 + 1] = 0.f;
      o_box[o * 4 + 2] = 0.f; o_box[o * 4 + 3] = 0.f;
      o_sc[o] = 0.0f;
      o_va[o] = 0.0f;
    }
  }
}

// =====================================================================
extern "C" void kernel_launch(void* const* d_in, const int* in_sizes, int n_in,
                              void* d_out, int out_size, void* d_ws, size_t ws_size,
                              hipStream_t stream) {
  (void)in_sizes; (void)n_in; (void)out_size; (void)ws_size;
  const float* feats  = (const float*)d_in[0];  // [B*N, 256]
  const float* Wm     = (const float*)d_in[1];  // [256, 4]
  const float* bias   = (const float*)d_in[2];  // [4]
  const float* priors = (const float*)d_in[3];  // [B*N, 4]
  const float* clog   = (const float*)d_in[4];  // [B*N, 81]

  float* out    = (float*)d_out;
  float* pred   = out;                                     // [B*N*4]
  float* o_lab  = pred  + (size_t)B_IMG * N_FEAT * 4;      // [B*100]
  float* o_box  = o_lab + (size_t)B_IMG * MAX_DETS;        // [B*100*4]
  float* o_sc   = o_box + (size_t)B_IMG * MAX_DETS * 4;    // [B*100]
  float* o_bi   = o_sc  + (size_t)B_IMG * MAX_DETS;        // [B*100]
  float* o_va   = o_bi  + (size_t)B_IMG * MAX_DETS;        // [B*100]

  // workspace carve-up (~520 KB)
  int*   hist = (int*)d_ws;                    // [B*NB]
  int*   thrb = hist + B_IMG * NB;             // [B]
  int*   cnt  = thrb + B_IMG;                  // [B]
  float* csc  = (float*)(cnt + B_IMG);         // [B*CAP]
  int*   cix  = (int*)(csc + B_IMG * CAP);     // [B*CAP]

  init_ws_kernel<<<(B_IMG * NB + 255) / 256, 256, 0, stream>>>(hist, thrb, cnt, csc, cix);
  gemm_decode_kernel<<<(B_IMG * N_FEAT) / 128, 256, 0, stream>>>(feats, Wm, bias, priors, pred);
  hist_kernel<<<B_IMG * (N_FEAT / 256), 256, 0, stream>>>(clog, pred, hist);
  thresh_kernel<<<1, 32, 0, stream>>>(hist, thrb);
  compact_kernel<<<B_IMG * (N_FEAT / 256), 256, 0, stream>>>(clog, pred, thrb, cnt, csc, cix);
  detect_kernel<<<B_IMG, 1024, 0, stream>>>(pred, csc, cix, o_lab, o_box, o_sc, o_bi, o_va);
}
